// ConvDecoderLayer_31301721653956
// MI455X (gfx1250) — matrix operand order repaired and checked by reading, not compiled, not run on hardware
//
#include <hip/hip_runtime.h>
#include <math.h>

typedef __bf16 bf16_t;
typedef __attribute__((ext_vector_type(16))) __bf16 v16bf;
typedef __attribute__((ext_vector_type(8)))  __bf16 v8bf;
typedef __attribute__((ext_vector_type(8)))  float  v8f;
typedef int v4i __attribute__((vector_size(16)));

#define BDIM 8
#define TDIM 2048
#define SDIM 2048
#define DDIM 512
#define BT_ROWS (BDIM * TDIM)   // 16384

// ---------------------------------------------------------------------------
// gfx1250 async global->LDS copy (probe-guarded; falls back to direct loads)
// ---------------------------------------------------------------------------
#if __has_builtin(__builtin_amdgcn_global_load_async_to_lds_b128)
#define HAVE_ASYNC_LDS 1
__device__ __forceinline__ void async_copy16(const void* g, void* l) {
  __builtin_amdgcn_global_load_async_to_lds_b128(
      (v4i*)g,
      (__attribute__((address_space(3))) v4i*)l, 0, 0);
}
__device__ __forceinline__ void wait_async0() {
#if __has_builtin(__builtin_amdgcn_s_wait_asynccnt)
  __builtin_amdgcn_s_wait_asynccnt(0);
#else
  asm volatile("s_wait_asynccnt 0x0" ::: "memory");
#endif
}
#endif

// ---------------------------------------------------------------------------
// WMMA helpers (CDNA5 gfx1250, wave32): D = A(16x32 bf16) * B(32x16 bf16) + C
// ---------------------------------------------------------------------------
__device__ __forceinline__ v8f wmma_bf16(v16bf a, v16bf b, v8f c) {
  return __builtin_amdgcn_wmma_f32_16x16x32_bf16(
      /*neg_a=*/false, a, /*neg_b=*/false, b,
      /*c_mod=*/(short)0, c, /*reuse_a=*/false, /*reuse_b=*/false);
}

// Build a v16bf fragment from two contiguous 8-element (16B) chunks.
__device__ __forceinline__ v16bf make_frag2(const bf16_t* p0, const bf16_t* p1) {
  v8bf a = *(const v8bf*)p0;
  v8bf b = *(const v8bf*)p1;
  v16bf f;
#pragma unroll
  for (int i = 0; i < 8; ++i) { f[i] = a[i]; f[8 + i] = b[i]; }
  return f;
}

// A fragment gathered from fp32 memory (converted to bf16 on the fly).
__device__ __forceinline__ v16bf make_frag_f32(const float* p) {
  v16bf f;
#pragma unroll
  for (int i = 0; i < 8; ++i) {
    f[i]     = (bf16_t)p[i];
    f[8 + i] = (bf16_t)p[16 + i];
  }
  return f;
}

// ---------------------------------------------------------------------------
// fp32 -> bf16 elementwise convert
// ---------------------------------------------------------------------------
__global__ void cvt_f32_bf16_kernel(const float* __restrict__ src,
                                    bf16_t* __restrict__ dst, int n) {
  int i = blockIdx.x * blockDim.x + threadIdx.x;
  if (i < n) dst[i] = (bf16_t)src[i];
}

// conv_w [O=1024][I=512][K=3] -> Wc bf16 [O][k_major=1536] with k = kw*512 + i
__global__ void pack_convw_kernel(const float* __restrict__ w,
                                  bf16_t* __restrict__ wc) {
  int i = blockIdx.x * 256 + threadIdx.x;
  if (i >= 1024 * 1536) return;
  int o  = i / 1536;
  int r  = i - o * 1536;
  int kw = r >> 9;        // 0..2
  int ci = r & 511;       // input channel
  wc[i] = (bf16_t)w[o * 1536 + ci * 3 + kw];
}

// ---------------------------------------------------------------------------
// Fused: causal conv (as WMMA GEMM, K=1536) + bias + GLU + residual + LN1
// grid = BT/16 blocks (16 t-rows of one batch), block = 256 (8 waves)
// ---------------------------------------------------------------------------
__global__ void __launch_bounds__(256) convln_kernel(
    const bf16_t* __restrict__ xb, const bf16_t* __restrict__ wc,
    const float* __restrict__ x, const float* __restrict__ conv_b,
    const float* __restrict__ ln_w, const float* __restrict__ ln_b,
    float* __restrict__ h32, bf16_t* __restrict__ hb) {
  __shared__ float sc[16][1024];   // 64 KB conv-output tile (a | g)

  const int tid  = threadIdx.x;
  const int lane = tid & 31;
  const int wv   = tid >> 5;
  const int hi   = lane >> 4;
  const int lw   = lane & 15;

  const int m0 = blockIdx.x * 16;
  const int b  = m0 / TDIM;
  const int t0 = m0 % TDIM;
  const bf16_t* xbb = xb + (size_t)b * TDIM * DDIM;

  v8f acc[8];
#pragma unroll
  for (int it = 0; it < 8; ++it) acc[it] = {};

  const int t = t0 + lw;                 // A-fragment row for this lane
  for (int kk = 0; kk < 1536; kk += 32) {
    const int kw  = kk >> 9;
    const int i0  = kk & 511;
    const int src = t - 2 + kw;          // causal left pad by K-1 = 2

    v16bf af;
    if (src >= 0) {
      const bf16_t* p = xbb + (size_t)src * DDIM + i0 + 8 * hi;
      af = make_frag2(p, p + 16);
    } else {
#pragma unroll
      for (int i = 0; i < 16; ++i) af[i] = (bf16_t)0.0f;
    }
    v16bf bfr[8];
#pragma unroll
    for (int it = 0; it < 8; ++it) {
      const int n0 = wv * 16 + it * 128;
      const bf16_t* q = wc + (size_t)(n0 + lw) * 1536 + kk + 8 * hi;
      bfr[it] = make_frag2(q, q + 16);
    }
#pragma unroll
    for (int it = 0; it < 8; ++it) acc[it] = wmma_bf16(af, bfr[it], acc[it]);
  }

  // park conv result (+bias) in LDS; C layout: lane holds col n, rows 8*hi+r
#pragma unroll
  for (int it = 0; it < 8; ++it) {
    const int n = wv * 16 + it * 128 + lw;
    const float cb = conv_b[n];
#pragma unroll
    for (int r = 0; r < 8; ++r) sc[8 * hi + r][n] = acc[it][r] + cb;
  }
  __syncthreads();

  // GLU + residual + LN1: each wave owns 2 rows
  for (int rr = wv * 2; rr < wv * 2 + 2; ++rr) {
    const int trow = t0 + rr;
    const float* xrow = x + ((size_t)b * TDIM + trow) * DDIM;
    float s = 0.f, sq = 0.f;
    float y[16];
#pragma unroll
    for (int i = 0; i < 16; ++i) {
      const int d = lane + 32 * i;
      const float a = sc[rr][d];
      const float g = sc[rr][d + 512];
      const float glu = a * (1.0f / (1.0f + __expf(-g)));
      const float v = glu + xrow[d];
      y[i] = v;
      s += v;
      sq += v * v;
    }
#pragma unroll
    for (int off = 16; off > 0; off >>= 1) {
      s  += __shfl_xor(s,  off, 32);
      sq += __shfl_xor(sq, off, 32);
    }
    const float mu  = s * (1.f / 512.f);
    const float var = sq * (1.f / 512.f) - mu * mu;
    const float rs  = rsqrtf(var + 1e-5f);
    float*  hrow  = h32 + ((size_t)b * TDIM + trow) * DDIM;
    bf16_t* hbrow = hb  + ((size_t)b * TDIM + trow) * DDIM;
#pragma unroll
    for (int i = 0; i < 16; ++i) {
      const int d = lane + 32 * i;
      const float o = (y[i] - mu) * rs * ln_w[d] + ln_b[d];
      hrow[d]  = o;
      hbrow[d] = (bf16_t)o;
    }
  }
}

// ---------------------------------------------------------------------------
// Generic 16x16-tile WMMA GEMM:  C[m,n] = sum_k A[m,k] * Bt[n,k] (+ bias[n])
// AFMT: 0 = bf16 A, 1 = fp32 A (converted per fragment)
// OMODE: 0 = bf16 out [M,512]; 1 = bf16 transposed per-batch [b][512][rows];
//        2 = fp32 out [M,512]
// grid = (M/16, 512/128), block = 256 (8 waves; wave w owns cols bIdx.y*128+16w)
// A panel (shared by all 8 waves) is staged global->LDS with async copies,
// double-buffered on ASYNCcnt; falls back to direct global loads if the
// async builtins are unavailable.
// ---------------------------------------------------------------------------
template <int AFMT, int OMODE>
__global__ void __launch_bounds__(256) gemm16_kernel(
    const void* __restrict__ Av, int lda,
    const bf16_t* __restrict__ Bt, int ldb, long long bt_bstride,
    int rows_per_batch, const float* __restrict__ bias,
    void* __restrict__ out, int Kdim) {
  const int tid  = threadIdx.x;
  const int lane = tid & 31;
  const int wv   = tid >> 5;
  const int hi   = lane >> 4;
  const int lw   = lane & 15;

  const int m0 = blockIdx.x * 16;
  const int n0 = blockIdx.y * 128 + wv * 16;
  const int bidx = m0 / rows_per_batch;
  const bf16_t* btb = Bt + (size_t)bidx * (size_t)bt_bstride;
  const bf16_t* Brow = btb + (size_t)(n0 + lw) * ldb + 8 * hi;

  v8f acc = {};

#if HAVE_ASYNC_LDS
  __shared__ char astage[2][2048];
  const int nsteps = Kdim >> 5;

  // issue stage 0
  if (AFMT == 0) {
    if (tid < 64) {
      const int row = tid >> 2, part = tid & 3;
      const bf16_t* g = (const bf16_t*)Av + (size_t)(m0 + row) * lda + part * 8;
      async_copy16(g, &astage[0][row * 64 + part * 16]);
    }
  } else {
    if (tid < 128) {
      const int row = tid >> 3, part = tid & 7;
      const float* g = (const float*)Av + (size_t)(m0 + row) * lda + part * 4;
      async_copy16(g, &astage[0][row * 128 + part * 16]);
    }
  }

  for (int i = 0; i < nsteps; ++i) {
    wait_async0();          // stage i resident in LDS (issuing waves)
    __syncthreads();        // publish to all waves; orders compute(i-1) first
    if (i + 1 < nsteps) {   // issue stage i+1 into the other buffer
      const int kk = (i + 1) << 5;
      char* dst = astage[(i + 1) & 1];
      if (AFMT == 0) {
        if (tid < 64) {
          const int row = tid >> 2, part = tid & 3;
          const bf16_t* g =
              (const bf16_t*)Av + (size_t)(m0 + row) * lda + kk + part * 8;
          async_copy16(g, dst + row * 64 + part * 16);
        }
      } else {
        if (tid < 128) {
          const int row = tid >> 3, part = tid & 7;
          const float* g =
              (const float*)Av + (size_t)(m0 + row) * lda + kk + part * 4;
          async_copy16(g, dst + row * 128 + part * 16);
        }
      }
    }
    // build A fragment from LDS, B fragment direct from global
    const char* abuf = astage[i & 1];
    v16bf a;
    if (AFMT == 0) {
      const bf16_t* p0 = (const bf16_t*)(abuf + lw * 64 + 16 * hi);
      a = make_frag2(p0, p0 + 16);
    } else {
      const float* p0 = (const float*)(abuf + lw * 128 + 32 * hi);
      a = make_frag_f32(p0);
    }
    const bf16_t* pb = Brow + (i << 5);
    acc = wmma_bf16(a, make_frag2(pb, pb + 16), acc);
  }
#else
  // fallback: direct global loads, double-buffered
  if (AFMT == 0) {
    const bf16_t* pa = (const bf16_t*)Av + (size_t)(m0 + lw) * lda + 8 * hi;
    v16bf a_cur = make_frag2(pa, pa + 16);
    v16bf b_cur = make_frag2(Brow, Brow + 16);
    for (int kk = 32; kk < Kdim; kk += 32) {
      v16bf a_nxt = make_frag2(pa + kk, pa + kk + 16);
      v16bf b_nxt = make_frag2(Brow + kk, Brow + kk + 16);
      acc = wmma_bf16(a_cur, b_cur, acc);
      a_cur = a_nxt;
      b_cur = b_nxt;
    }
    acc = wmma_bf16(a_cur, b_cur, acc);
  } else {
    const float* pa = (const float*)Av + (size_t)(m0 + lw) * lda + 8 * hi;
    v16bf a_cur = make_frag_f32(pa);
    v16bf b_cur = make_frag2(Brow, Brow + 16);
    for (int kk = 32; kk < Kdim; kk += 32) {
      v16bf a_nxt = make_frag_f32(pa + kk);
      v16bf b_nxt = make_frag2(Brow + kk, Brow + kk + 16);
      acc = wmma_bf16(a_cur, b_cur, acc);
      a_cur = a_nxt;
      b_cur = b_nxt;
    }
    acc = wmma_bf16(a_cur, b_cur, acc);
  }
#endif

  const int n = n0 + lw;
  const float bv = bias ? bias[n] : 0.0f;
#pragma unroll
  for (int r = 0; r < 8; ++r) {
    const int m = m0 + 8 * hi + r;
    const float v = acc[r] + bv;
    if (OMODE == 0) {
      ((bf16_t*)out)[(size_t)m * 512 + n] = (bf16_t)v;
    } else if (OMODE == 1) {
      const int bb = m / rows_per_batch;
      const int s  = m - bb * rows_per_batch;
      ((bf16_t*)out)[((size_t)bb * 512 + n) * rows_per_batch + s] = (bf16_t)v;
    } else {
      ((float*)out)[(size_t)m * 512 + n] = v;
    }
  }
}

// ---------------------------------------------------------------------------
// Fused scores + softmax: 16 Q rows x full S = 2048 per block.
// Q A-frags cached in VGPRs; K rows loaded directly as B-frags (Q.K^T shape).
// grid = BT/16, block = 256 (8 waves); wave w owns cols it*128 + 16w.
// ---------------------------------------------------------------------------
__global__ void __launch_bounds__(256) attn_kernel(
    const bf16_t* __restrict__ Qb, const bf16_t* __restrict__ Kb,
    float* __restrict__ attn_out) {
  __shared__ float redbuf[8][16];

  const int tid  = threadIdx.x;
  const int lane = tid & 31;
  const int wv   = tid >> 5;
  const int hi   = lane >> 4;
  const int lw   = lane & 15;

  const int m0 = blockIdx.x * 16;
  const int b  = m0 / TDIM;
  const bf16_t* kbb = Kb + (size_t)b * SDIM * DDIM;

  // cache the 16x512 Q tile as 16 A-fragments
  v16bf qa[16];
  {
    const bf16_t* qrow = Qb + ((size_t)m0 + lw) * DDIM;
#pragma unroll
    for (int kk = 0; kk < 16; ++kk) {
      const bf16_t* p = qrow + kk * 32 + 8 * hi;
      qa[kk] = make_frag2(p, p + 16);
    }
  }

  const float scale = 0.044194173824159216f;  // 1/sqrt(512)
  v8f acc[16];
#pragma unroll
  for (int it = 0; it < 16; ++it) {
    const bf16_t* krow = kbb + (size_t)(it * 128 + wv * 16 + lw) * DDIM;
    // prefetch the next K panel (gfx1250 global_prefetch path)
    if (it + 1 < 16) {
      const bf16_t* knext = kbb + (size_t)((it + 1) * 128 + wv * 16 + lw) * DDIM;
      __builtin_prefetch(knext, 0, 1);
      __builtin_prefetch(knext + 256, 0, 1);
    }
    v8f c = {};
    const bf16_t* pb = krow + 8 * hi;
    v16bf b_cur = make_frag2(pb, pb + 16);
#pragma unroll
    for (int kk = 0; kk < 15; ++kk) {
      v16bf b_nxt = make_frag2(pb + (kk + 1) * 32, pb + (kk + 1) * 32 + 16);
      c = wmma_bf16(qa[kk], b_cur, c);
      b_cur = b_nxt;
    }
    c = wmma_bf16(qa[15], b_cur, c);
#pragma unroll
    for (int r = 0; r < 8; ++r) c[r] *= scale;
    acc[it] = c;
  }

  // --- row max ---
  float rmax[8];
#pragma unroll
  for (int r = 0; r < 8; ++r) {
    float m = acc[0][r];
#pragma unroll
    for (int it = 1; it < 16; ++it) m = fmaxf(m, acc[it][r]);
#pragma unroll
    for (int off = 1; off < 16; off <<= 1) m = fmaxf(m, __shfl_xor(m, off, 32));
    rmax[r] = m;
  }
  if (lw == 0) {
#pragma unroll
    for (int r = 0; r < 8; ++r) redbuf[wv][8 * hi + r] = rmax[r];
  }
  __syncthreads();
#pragma unroll
  for (int r = 0; r < 8; ++r) {
    float m = redbuf[0][8 * hi + r];
#pragma unroll
    for (int w = 1; w < 8; ++w) m = fmaxf(m, redbuf[w][8 * hi + r]);
    rmax[r] = m;
  }
  __syncthreads();

  // --- exp + row sum ---
  float rsum[8];
#pragma unroll
  for (int r = 0; r < 8; ++r) rsum[r] = 0.f;
#pragma unroll
  for (int it = 0; it < 16; ++it) {
#pragma unroll
    for (int r = 0; r < 8; ++r) {
      const float e = __expf(acc[it][r] - rmax[r]);
      acc[it][r] = e;
      rsum[r] += e;
    }
  }
#pragma unroll
  for (int r = 0; r < 8; ++r) {
    float s = rsum[r];
#pragma unroll
    for (int off = 1; off < 16; off <<= 1) s += __shfl_xor(s, off, 32);
    rsum[r] = s;
  }
  if (lw == 0) {
#pragma unroll
    for (int r = 0; r < 8; ++r) redbuf[wv][8 * hi + r] = rsum[r];
  }
  __syncthreads();
#pragma unroll
  for (int r = 0; r < 8; ++r) {
    float s = 0.f;
#pragma unroll
    for (int w = 0; w < 8; ++w) s += redbuf[w][8 * hi + r];
    rsum[r] = 1.0f / s;
  }

  // --- write normalized attention (fp32, straight into d_out) ---
#pragma unroll
  for (int it = 0; it < 16; ++it) {
    const int scol = it * 128 + wv * 16 + lw;
#pragma unroll
    for (int r = 0; r < 8; ++r) {
      const int row = m0 + 8 * hi + r;
      attn_out[(size_t)row * SDIM + scol] = acc[it][r] * rsum[r];
    }
  }
}

// ---------------------------------------------------------------------------
// Fused residual + LN2: out = LN(h + ctx_o). grid = BT rows, block = 256.
// ---------------------------------------------------------------------------
__global__ void __launch_bounds__(256) ln2_kernel(
    const float* __restrict__ h32, const float* __restrict__ ctxo,
    const float* __restrict__ w, const float* __restrict__ bws,
    float* __restrict__ out) {
  __shared__ float rbuf[16];
  const int m    = blockIdx.x;
  const int tid  = threadIdx.x;
  const int lane = tid & 31;
  const int wv   = tid >> 5;

  const float* hr = h32 + (size_t)m * 512;
  const float* cr = ctxo + (size_t)m * 512;
  const float y0 = hr[tid] + cr[tid];
  const float y1 = hr[tid + 256] + cr[tid + 256];
  float s = y0 + y1, sq = y0 * y0 + y1 * y1;
#pragma unroll
  for (int off = 16; off > 0; off >>= 1) {
    s  += __shfl_xor(s,  off, 32);
    sq += __shfl_xor(sq, off, 32);
  }
  if (lane == 0) { rbuf[wv] = s; rbuf[8 + wv] = sq; }
  __syncthreads();
  s = 0.f; sq = 0.f;
#pragma unroll
  for (int i = 0; i < 8; ++i) { s += rbuf[i]; sq += rbuf[8 + i]; }
  const float mu  = s * (1.f / 512.f);
  const float var = sq * (1.f / 512.f) - mu * mu;
  const float rs  = rsqrtf(var + 1e-5f);
  out[(size_t)m * 512 + tid]       = (y0 - mu) * rs * w[tid] + bws[tid];
  out[(size_t)m * 512 + tid + 256] = (y1 - mu) * rs * w[tid + 256] + bws[tid + 256];
}

// ---------------------------------------------------------------------------
// Host driver
// ---------------------------------------------------------------------------
extern "C" void kernel_launch(void* const* d_in, const int* in_sizes, int n_in,
                              void* d_out, int out_size, void* d_ws,
                              size_t ws_size, hipStream_t stream) {
  (void)in_sizes; (void)n_in; (void)out_size; (void)ws_size;

  const float* x      = (const float*)d_in[0];
  const float* enc    = (const float*)d_in[1];
  const float* conv_w = (const float*)d_in[2];
  const float* conv_b = (const float*)d_in[3];
  const float* ln1_w  = (const float*)d_in[4];
  const float* ln1_b  = (const float*)d_in[5];
  const float* Wq     = (const float*)d_in[6];
  const float* bq     = (const float*)d_in[7];
  const float* Wk     = (const float*)d_in[8];
  const float* bk     = (const float*)d_in[9];
  const float* Wv     = (const float*)d_in[10];
  const float* bv     = (const float*)d_in[11];
  const float* Wo     = (const float*)d_in[12];
  const float* bo     = (const float*)d_in[13];
  const float* ln2_w  = (const float*)d_in[14];
  const float* ln2_b  = (const float*)d_in[15];

  char* ws = (char*)d_ws;
  const size_t MB = 1ull << 20;
  bf16_t* xb   = (bf16_t*)(ws);             // 16 MiB
  bf16_t* encb = (bf16_t*)(ws + 16 * MB);   // 16 MiB
  bf16_t* hb   = (bf16_t*)(ws + 32 * MB);   // 16 MiB
  bf16_t* Qb   = (bf16_t*)(ws + 48 * MB);   // 16 MiB (reused as ctxb)
  bf16_t* Kb   = (bf16_t*)(ws + 64 * MB);   // 16 MiB
  bf16_t* Vt   = (bf16_t*)(ws + 80 * MB);   // 16 MiB (transposed V [b][d][s])
  bf16_t* ctxb = Qb;                        // reuse after attention
  float*  ctxo = (float*)(ws + 64 * MB);    // 32 MiB, overlays dead Kb+Vt
  float*  h32  = (float*)(ws + 96 * MB);    // 32 MiB
  bf16_t* Wc   = (bf16_t*)(ws + 128 * MB);  // 3 MiB
  bf16_t* Wqb  = (bf16_t*)(ws + 131 * MB);  // 0.5 MiB each
  bf16_t* Wkb  = (bf16_t*)(ws + 131 * MB + 512 * 1024);
  bf16_t* Wvb  = (bf16_t*)(ws + 132 * MB);
  bf16_t* Wob  = (bf16_t*)(ws + 132 * MB + 512 * 1024);

  float* out_main = (float*)d_out;
  float* attn_out = out_main + (size_t)BT_ROWS * DDIM;

  const int NXE = BT_ROWS * DDIM;  // 8388608
  const int NW  = DDIM * DDIM;     // 262144

  // 1) converts / packs
  cvt_f32_bf16_kernel<<<(NXE + 255) / 256, 256, 0, stream>>>(x, xb, NXE);
  cvt_f32_bf16_kernel<<<(NXE + 255) / 256, 256, 0, stream>>>(enc, encb, NXE);
  cvt_f32_bf16_kernel<<<(NW + 255) / 256, 256, 0, stream>>>(Wq, Wqb, NW);
  cvt_f32_bf16_kernel<<<(NW + 255) / 256, 256, 0, stream>>>(Wk, Wkb, NW);
  cvt_f32_bf16_kernel<<<(NW + 255) / 256, 256, 0, stream>>>(Wv, Wvb, NW);
  cvt_f32_bf16_kernel<<<(NW + 255) / 256, 256, 0, stream>>>(Wo, Wob, NW);
  pack_convw_kernel<<<(1024 * 1536 + 255) / 256, 256, 0, stream>>>(conv_w, Wc);

  // 2) conv + GLU + residual + LN1  ->  h32 (fp32), hb (bf16)
  convln_kernel<<<BT_ROWS / 16, 256, 0, stream>>>(xb, Wc, x, conv_b, ln1_w,
                                                  ln1_b, h32, hb);

  // 3) projections
  dim3 gp(BT_ROWS / 16, 4);
  gemm16_kernel<0, 0><<<gp, 256, 0, stream>>>(hb, DDIM, Wqb, DDIM, 0LL, TDIM,
                                              bq, Qb, DDIM);
  gemm16_kernel<0, 0><<<gp, 256, 0, stream>>>(encb, DDIM, Wkb, DDIM, 0LL, SDIM,
                                              bk, Kb, DDIM);
  gemm16_kernel<0, 1><<<gp, 256, 0, stream>>>(encb, DDIM, Wvb, DDIM, 0LL, SDIM,
                                              bv, Vt, DDIM);

  // 4) scores + softmax -> attn (fp32, second output)
  attn_kernel<<<BT_ROWS / 16, 256, 0, stream>>>(Qb, Kb, attn_out);

  // 5) ctx = attn @ V  (A fp32 from d_out, B from transposed V)
  gemm16_kernel<1, 0><<<gp, 256, 0, stream>>>(
      attn_out, SDIM, Vt, SDIM, (long long)DDIM * SDIM, TDIM, nullptr, ctxb,
      SDIM);

  // 6) ctx @ Wo^T + bo -> ctxo (fp32)
  gemm16_kernel<0, 2><<<gp, 256, 0, stream>>>(ctxb, DDIM, Wob, DDIM, 0LL, TDIM,
                                              bo, ctxo, DDIM);

  // 7) out = LN2(h + ctxo)
  ln2_kernel<<<BT_ROWS, 256, 0, stream>>>(h32, ctxo, ln2_w, ln2_b, out_main);
}
